// EDBLoss_3676492005810
// MI455X (gfx1250) — matrix operand order repaired and checked by reading, not compiled, not run on hardware
//
#include <hip/hip_runtime.h>
#include <stdint.h>

// ---------------------------------------------------------------------------
// MI455X / gfx1250 implementation of the easy-positive distance-border loss.
//  K1: split fp32 -> (bf16 hi, bf16 lo) + row squared norms
//  K2: WMMA bf16x3 GEMM (hi*hi + hi*lo + lo*hi) fused with dist = sqrt(...)
//      LDS panels double-buffered and filled with GLOBAL_LOAD_ASYNC_TO_LDS_B128
//      (ASYNCcnt-pipelined DMA, no VGPR round trip).
//  K3: per-row top-10 selection via packed u64 keys + LDS merge tree
//  K4: final deterministic reduction of per-row losses -> 3 scalars
// Workspace: Xhi(16MB) + Xlo(16MB) + dist(64MB) + sq + rowloss  (~96MB)
// ---------------------------------------------------------------------------

typedef __attribute__((ext_vector_type(16))) __bf16 v16bf;
typedef __attribute__((ext_vector_type(8)))  float  v8f;

union FragBF {
    uint4 q[2];
    v16bf v;
};

// ---- gfx1250 async global->LDS path (ASYNCcnt), with safe fallback --------
#if defined(__has_builtin)
#if __has_builtin(__builtin_amdgcn_global_load_async_to_lds_b128)
#define HAVE_ASYNC_LDS 1
#endif
#endif

#ifdef HAVE_ASYNC_LDS
// Builtin signature (from hipcc diagnostic): params are pointers to
// 'int __attribute__((vector_size(16)))' — global src, LDS dst.
typedef int async_v4i __attribute__((vector_size(16)));
typedef __attribute__((address_space(1))) async_v4i* gbl_v4i_p;
typedef __attribute__((address_space(3))) async_v4i* lds_v4i_p;
#define ASYNC_B128(gsrc, ldst)                                                  \
    __builtin_amdgcn_global_load_async_to_lds_b128(                             \
        (gbl_v4i_p)(gsrc), (lds_v4i_p)(ldst), 0, 0)
#if __has_builtin(__builtin_amdgcn_s_wait_asynccnt)
#define WAIT_ASYNC(n) __builtin_amdgcn_s_wait_asynccnt(n)
#else
#define WAIT_ASYNC(n) asm volatile("s_wait_asynccnt %0" ::"i"(n) : "memory")
#endif
#else
#define ASYNC_B128(gsrc, ldst) (*(uint4*)(ldst) = *(const uint4*)(gsrc))
#define WAIT_ASYNC(n)
#endif

static __device__ __forceinline__ unsigned short f32_to_bf16_rne(float f) {
    unsigned int u = __float_as_uint(f);
    unsigned int r = u + 0x7FFFu + ((u >> 16) & 1u);
    return (unsigned short)(r >> 16);
}
static __device__ __forceinline__ float bf16_bits_to_f32(unsigned short h) {
    return __uint_as_float(((unsigned int)h) << 16);
}
static __device__ __forceinline__ unsigned int pack_hi2(float x, float y) {
    return (unsigned int)f32_to_bf16_rne(x) | ((unsigned int)f32_to_bf16_rne(y) << 16);
}
static __device__ __forceinline__ unsigned int pack_lo2(float x, float y) {
    float rx = x - bf16_bits_to_f32(f32_to_bf16_rne(x));
    float ry = y - bf16_bits_to_f32(f32_to_bf16_rne(y));
    return (unsigned int)f32_to_bf16_rne(rx) | ((unsigned int)f32_to_bf16_rne(ry) << 16);
}

// --------------------------------------------------------------------------
// K1: one block per row. Split into hi/lo bf16 and reduce squared norm.
// --------------------------------------------------------------------------
__global__ void __launch_bounds__(256)
k_split_sq(const float* __restrict__ X, unsigned short* __restrict__ Xhi,
           unsigned short* __restrict__ Xlo, float* __restrict__ sq, int C) {
    __shared__ float sm[256];
    const int row = blockIdx.x;
    const int t = threadIdx.x;
    const float* xr = X + (size_t)row * C;
    float acc = 0.f;
    for (int j8 = t; j8 < (C >> 3); j8 += 256) {
        const float4* p = (const float4*)(xr + j8 * 8);
        float4 a = p[0], b = p[1];
        uint4 hq = make_uint4(pack_hi2(a.x, a.y), pack_hi2(a.z, a.w),
                              pack_hi2(b.x, b.y), pack_hi2(b.z, b.w));
        uint4 lq = make_uint4(pack_lo2(a.x, a.y), pack_lo2(a.z, a.w),
                              pack_lo2(b.x, b.y), pack_lo2(b.z, b.w));
        *(uint4*)(Xhi + (size_t)row * C + j8 * 8) = hq;
        *(uint4*)(Xlo + (size_t)row * C + j8 * 8) = lq;
        acc += a.x * a.x + a.y * a.y + a.z * a.z + a.w * a.w;
        acc += b.x * b.x + b.y * b.y + b.z * b.z + b.w * b.w;
    }
    sm[t] = acc;
    __syncthreads();
    for (int s = 128; s > 0; s >>= 1) {
        if (t < s) sm[t] += sm[t + s];
        __syncthreads();
    }
    if (t == 0) sq[row] = sm[0];
}

// --------------------------------------------------------------------------
// K2: 64x64 tile per 128-thread (4-wave) block; wave owns a 32x32 quadrant
//     as 2x2 WMMA 16x16 tiles. bf16x3 split product, fused sqrt-dist store.
//     LDS panels double-buffered; filled by async global->LDS DMA.
// --------------------------------------------------------------------------
#define TK 32
#define LS 56   // LDS row pitch in ushorts: 112B, 16B aligned, conflict-free

__global__ void __launch_bounds__(128)
k_gemm_dist(const unsigned short* __restrict__ Xhi,
            const unsigned short* __restrict__ Xlo,
            const float* __restrict__ sq,
            float* __restrict__ dist, int N, int C) {
    __shared__ unsigned short Ah[2][64 * LS], Al[2][64 * LS];
    __shared__ unsigned short Bh[2][64 * LS], Bl[2][64 * LS];

    const int t    = threadIdx.x;
    const int lane = t & 31;
    const int w    = t >> 5;
    const int half = lane >> 4;   // 0: lanes 0-15, 1: lanes 16-31
    const int l15  = lane & 15;
    const int mwb  = (w >> 1) * 32;   // wave M base in block tile
    const int nwb  = (w & 1) * 32;    // wave N base in block tile
    const int rowBase = blockIdx.y * 64;
    const int colBase = blockIdx.x * 64;

    const int lrow = t >> 1;          // 0..63 : staging row per thread
    const int lk   = (t & 1) * 16;    // 0 or 16 within K chunk

    const size_t gaOff = (size_t)(rowBase + lrow) * C + lk;   // A panel source
    const size_t gbOff = (size_t)(colBase + lrow) * C + lk;   // B panel source
    const int    lOff  = lrow * LS + lk;

    v8f acc[2][2] = {};

    // issue one K-chunk's 8 async 16B packets for this thread
    auto issue = [&](int buf, int kb) {
        ASYNC_B128(Xhi + gaOff + kb,     &Ah[buf][lOff]);
        ASYNC_B128(Xhi + gaOff + kb + 8, &Ah[buf][lOff + 8]);
        ASYNC_B128(Xlo + gaOff + kb,     &Al[buf][lOff]);
        ASYNC_B128(Xlo + gaOff + kb + 8, &Al[buf][lOff + 8]);
        ASYNC_B128(Xhi + gbOff + kb,     &Bh[buf][lOff]);
        ASYNC_B128(Xhi + gbOff + kb + 8, &Bh[buf][lOff + 8]);
        ASYNC_B128(Xlo + gbOff + kb,     &Bl[buf][lOff]);
        ASYNC_B128(Xlo + gbOff + kb + 8, &Bl[buf][lOff + 8]);
    };

    const int nChunks = C / TK;
    issue(0, 0);
    for (int ic = 0; ic < nChunks; ++ic) {
        const int buf = ic & 1;
        if (ic + 1 < nChunks) {
            issue(buf ^ 1, (ic + 1) * TK);   // next chunk in flight behind WMMA
            WAIT_ASYNC(8);                   // current chunk's 8 packets landed
        } else {
            WAIT_ASYNC(0);
        }
        __syncthreads();

        // A frag lane<16: M=lane, K 0..7 & 16..23 ; lane>=16: K 8..15 & 24..31
        // B frag lane<16: N=lane, K 0..7 & 16..23 ; lane>=16: K 8..15 & 24..31
#pragma unroll
        for (int mt = 0; mt < 2; ++mt) {
            FragBF ahi, alo;
            const int ar = (mwb + mt * 16 + l15) * LS;
            ahi.q[0] = *(const uint4*)&Ah[buf][ar + half * 8];
            ahi.q[1] = *(const uint4*)&Ah[buf][ar + 16 + half * 8];
            alo.q[0] = *(const uint4*)&Al[buf][ar + half * 8];
            alo.q[1] = *(const uint4*)&Al[buf][ar + 16 + half * 8];
#pragma unroll
            for (int nt = 0; nt < 2; ++nt) {
                FragBF bhi, blo;
                const int br = (nwb + nt * 16 + l15) * LS + half * 8;
                bhi.q[0] = *(const uint4*)&Bh[buf][br];
                bhi.q[1] = *(const uint4*)&Bh[buf][br + 16];
                blo.q[0] = *(const uint4*)&Bl[buf][br];
                blo.q[1] = *(const uint4*)&Bl[buf][br + 16];
                v8f c = acc[mt][nt];
                c = __builtin_amdgcn_wmma_f32_16x16x32_bf16(false, ahi.v, false, bhi.v,
                                                            (short)0, c, false, false);
                c = __builtin_amdgcn_wmma_f32_16x16x32_bf16(false, ahi.v, false, blo.v,
                                                            (short)0, c, false, false);
                c = __builtin_amdgcn_wmma_f32_16x16x32_bf16(false, alo.v, false, bhi.v,
                                                            (short)0, c, false, false);
                acc[mt][nt] = c;
            }
        }
        __syncthreads();   // all waves done reading buf before it is refilled
    }

    // ---- epilogue: dist = sqrt(max(sq_i + sq_j - 2*dot, 1e-12))
#pragma unroll
    for (int mt = 0; mt < 2; ++mt) {
#pragma unroll
        for (int nt = 0; nt < 2; ++nt) {
            const int gi0 = rowBase + mwb + mt * 16 + half * 8;
            const int gj  = colBase + nwb + nt * 16 + l15;
            const float sqj = sq[gj];
#pragma unroll
            for (int v = 0; v < 8; ++v) {
                const int gi = gi0 + v;
                float d2 = sq[gi] + sqj - 2.0f * acc[mt][nt][v];
                d2 = fmaxf(d2, 1e-12f);
                dist[(size_t)gi * N + gj] = sqrtf(d2);
            }
        }
    }
}

// --------------------------------------------------------------------------
// K3: one block per row. Top-10 via packed (dist_bits<<32 | j) keys:
//     per-thread register top-10, then LDS ping-pong merge tree.
// --------------------------------------------------------------------------
__global__ void __launch_bounds__(256)
k_rowselect(const float* __restrict__ dist, const long long* __restrict__ labels,
            float* __restrict__ rowloss, int N) {
    __shared__ unsigned long long cand[2][256 * 10];
    __shared__ float sSum[256];
    __shared__ int   sCnt[256];

    const int row = blockIdx.x;
    const int t = threadIdx.x;
    const long long lab = labels[row];
    const float* dr = dist + (size_t)row * N;

    unsigned long long top[10];
#pragma unroll
    for (int q = 0; q < 10; ++q) top[q] = 0xFFFFFFFFFFFFFFFFull;

    float psum = 0.f;
    int   pcnt = 0;
    for (int j = t; j < N; j += 256) {
        const float d = dr[j];
        const bool same = (labels[j] == lab);
        psum += same ? d : 0.f;
        pcnt += same ? 1 : 0;
        unsigned long long key =
            (((unsigned long long)__float_as_uint(d)) << 32) | (unsigned int)j;
        if (key < top[9]) {
#pragma unroll
            for (int q = 0; q < 10; ++q) {   // bubble insert, constant indices
                const unsigned long long mn = key < top[q] ? key : top[q];
                const unsigned long long mx = key < top[q] ? top[q] : key;
                top[q] = mn;
                key = mx;
            }
        }
    }
#pragma unroll
    for (int q = 0; q < 10; ++q) cand[0][t * 10 + q] = top[q];
    sSum[t] = psum;
    sCnt[t] = pcnt;
    __syncthreads();

    int phase = 0;
    for (int s = 128; s > 0; s >>= 1) {
        if (t < s) {
            const unsigned long long* a = &cand[phase][t * 10];
            const unsigned long long* b = &cand[phase][(t + s) * 10];
            unsigned long long* o = &cand[phase ^ 1][t * 10];
            int ia = 0, ib = 0;
#pragma unroll
            for (int k = 0; k < 10; ++k) {
                const unsigned long long av = a[ia], bv = b[ib];
                if (av <= bv) { o[k] = av; ++ia; }
                else          { o[k] = bv; ++ib; }
            }
            sSum[t] += sSum[t + s];
            sCnt[t] += sCnt[t + s];
        }
        phase ^= 1;
        __syncthreads();
    }

    if (t == 0) {
        const unsigned long long* g = &cand[phase][0];
        const float border = __uint_as_float((unsigned int)(g[9] >> 32));
        float aeSum = 0.f, anSum = 0.f, apExSum = 0.f;
        int aeCnt = 0, anCnt = 0, apExCnt = 0;
        for (int q = 0; q < 10; ++q) {
            const unsigned long long key = g[q];
            const float d = __uint_as_float((unsigned int)(key >> 32));
            const int j = (int)(unsigned int)(key & 0xFFFFFFFFu);
            if (labels[j] == lab) {
                apExSum += d; apExCnt += 1;
                aeSum += fmaxf(0.5f - d, 0.f); aeCnt += 1;          // MARGIN2
            } else {
                anSum += fmaxf(border + 1.3f - d, 0.f); anCnt += 1; // MARGIN1
            }
        }
        const float tSum = sSum[0] - apExSum;       // tail positives: sum(d)
        const int   tCnt = sCnt[0] - apExCnt;       // tail positives: count
        rowloss[row * 3 + 0] = tCnt > 0 ? (tSum - border * (float)tCnt) / (float)tCnt : 0.f;
        rowloss[row * 3 + 1] = anCnt > 0 ? anSum / (float)anCnt : 0.f;
        rowloss[row * 3 + 2] = aeCnt > 0 ? aeSum / (float)aeCnt : 0.f;
    }
}

// --------------------------------------------------------------------------
// K4: deterministic final reduction of per-row losses -> 3 output scalars.
// --------------------------------------------------------------------------
__global__ void __launch_bounds__(256)
k_final(const float* __restrict__ rowloss, float* __restrict__ out, int N) {
    __shared__ float s0[256], s1[256], s2[256];
    const int t = threadIdx.x;
    float a = 0.f, b = 0.f, c = 0.f;
    for (int r = t; r < N; r += 256) {
        a += rowloss[r * 3 + 0];
        b += rowloss[r * 3 + 1];
        c += rowloss[r * 3 + 2];
    }
    s0[t] = a; s1[t] = b; s2[t] = c;
    __syncthreads();
    for (int s = 128; s > 0; s >>= 1) {
        if (t < s) { s0[t] += s0[t + s]; s1[t] += s1[t + s]; s2[t] += s2[t + s]; }
        __syncthreads();
    }
    if (t == 0) {
        const float inv = 1.0f / (float)N;
        out[0] = s0[0] * inv;
        out[1] = s1[0] * inv;
        out[2] = s2[0] * inv;
    }
}

extern "C" void kernel_launch(void* const* d_in, const int* in_sizes, int n_in,
                              void* d_out, int out_size, void* d_ws, size_t ws_size,
                              hipStream_t stream) {
    (void)n_in; (void)out_size; (void)ws_size;
    const float*     X      = (const float*)d_in[0];
    const long long* labels = (const long long*)d_in[1];
    const int N = in_sizes[1];        // 4096 rows
    const int C = in_sizes[0] / N;    // 2048 features

    char* ws = (char*)d_ws;
    size_t off = 0;
    unsigned short* Xhi = (unsigned short*)(ws + off); off += (size_t)N * C * 2;
    unsigned short* Xlo = (unsigned short*)(ws + off); off += (size_t)N * C * 2;
    float* dist    = (float*)(ws + off); off += (size_t)N * N * 4;
    float* sq      = (float*)(ws + off); off += (size_t)N * 4;
    float* rowloss = (float*)(ws + off); off += (size_t)N * 3 * 4;
    float* out = (float*)d_out;

    k_split_sq<<<N, 256, 0, stream>>>(X, Xhi, Xlo, sq, C);
    dim3 g2(N / 64, N / 64);
    k_gemm_dist<<<g2, 128, 0, stream>>>(Xhi, Xlo, sq, dist, N, C);
    k_rowselect<<<N, 256, 0, stream>>>(dist, labels, rowloss, N);
    k_final<<<1, 256, 0, stream>>>(rowloss, out, N);
}
